// raw_BiLSTM_41747082117649
// MI455X (gfx1250) — matrix operand order repaired and checked by reading, not compiled, not run on hardware
//
#include <hip/hip_runtime.h>
#include <hip/hip_bf16.h>

// ---------------------------------------------------------------------------
// BiLSTM (V=256, H=512, D=2, B=256, W=256) for gfx1250 (MI455X).
//  - bf16 WMMA (v_wmma_f32_16x16x32_bf16), fp32 accumulate; weights L2-resident.
//  - Fused gates-GEMM + LSTM cell (i/f/g/o in same lane -> register epilogue).
//  - W tiles fetched by Tensor Data Mover: ONE tensor_load_to_lds per K-chunk
//    (3D tile: K=64 x rows=64 x gates=4, dim1_stride=512*768, LDS pad to LDA=80),
//    issued by wave 0, synced with s_wait_tensorcnt + workgroup barrier.
//  - A tiles via GLOBAL_LOAD_ASYNC_TO_LDS_B128 (ASYNCcnt), double-buffered LDS.
// ---------------------------------------------------------------------------

typedef __bf16 v16bf __attribute__((ext_vector_type(16)));
typedef __bf16 v8bf  __attribute__((ext_vector_type(8)));
typedef float  v8f   __attribute__((ext_vector_type(8)));
typedef unsigned v4u_ __attribute__((ext_vector_type(4)));
typedef int      v4i_ __attribute__((ext_vector_type(4)));
typedef int      v8i_ __attribute__((ext_vector_type(8)));

#define Vv    256
#define Hh    512
#define Gg    2048      // 4*H
#define Bb    256
#define Wlen  256
#define Tt    255       // W-1 steps per direction
#define KCAT  768       // V + H concatenated K
#define KC    64        // K chunk staged in LDS
#define LDA   80        // LDS row stride (elems), 160B

// ---- workspace layout (bytes) ----
#define OFF_WCAT   0u                          // 2dir*2dep*2048*768 bf16 = 12582912
#define OFF_WFC    12582912u                   // 2*2*256*512 bf16       =  1048576
#define OFF_BSUM   13631488u                   // 2*2*2048 f32           =    32768 (65536 reserved)
#define OFF_C      13697024u                   // 2*2*256*512 f32        =  2097152
#define OFF_ACAT   15794176u                   // 2*2*256*768 bf16       =  1572864

__device__ __forceinline__ float sigf(float x) { return 1.0f / (1.0f + __expf(-x)); }

// CDNA5 async DMA: global -> LDS, 16B per lane, tracked by ASYNCcnt.
__device__ __forceinline__ void async_b128(unsigned lds_byte_off, const void* g) {
    asm volatile("global_load_async_to_lds_b128 %0, %1, off"
                 :: "v"(lds_byte_off), "v"((unsigned long long)(uintptr_t)g)
                 : "memory");
}
__device__ __forceinline__ void wait_async0() {
    asm volatile("s_wait_asynccnt 0x0" ::: "memory");
}
__device__ __forceinline__ unsigned lds_off(const void* p) {
    return (unsigned)(uintptr_t)p;   // generic LDS addr: low 32 bits = LDS offset
}

// ---------------------------------------------------------------------------
// TDM descriptor (CDNA5 ISA): load a 3D tile
//   (dim0 = KC elems of 2B, dim1 = 64 rows, dim2 = 4 gate blocks)
// into LDS with pad_enable: +8 DWORDs after every 32 DWORDs -> row stride 80
// elems (== LDA). dim0_stride = KCAT rows, dim1_stride = Hh*KCAT between gates.
// ---------------------------------------------------------------------------
__device__ __forceinline__ void tdm_load_w(unsigned lds_addr, const void* gtile)
{
    const unsigned long long ga = (unsigned long long)(uintptr_t)gtile;
    v4u_ g0;
    g0[0] = 1u;                                            // count=1 (user D#)
    g0[1] = lds_addr;                                      // lds_addr [63:32]
    g0[2] = (unsigned)ga;                                  // global_addr[31:0]
    g0[3] = (unsigned)((ga >> 32) & 0x01ffffffu)           // global_addr[56:32]
          | (2u << 30);                                    // type=2 ("image")
    v8i_ g1;
    g1[0] = (int)((1u << 16)                               // data_size=1 -> 2B
                | (1u << 20)                               // pad_enable
                | (4u << 22)                               // pad_interval: 32 DW
                | (7u << 25));                             // pad_amount: 8 DW
    g1[1] = (int)((unsigned)KC << 16);                     // tensor_dim0 = 64
    g1[2] = (int)(64u << 16);                              // tensor_dim1 = 64
    g1[3] = (int)((unsigned)KC << 16);                     // tile_dim0 = 64
    g1[4] = (int)(64u | (4u << 16));                       // tile_dim1=64, tile_dim2=4
    g1[5] = (int)KCAT;                                     // tensor_dim0_stride = 768
    g1[6] = (int)(((unsigned)(Hh * KCAT) & 0xffffu) << 16);// dim1_stride[15:0] (=0)
    g1[7] = (int)((unsigned)(Hh * KCAT) >> 16);            // dim1_stride[47:16]
    v4i_ g2;
    g2[0] = 4;                                             // tensor_dim2 = 4
    g2[1] = 0; g2[2] = 0; g2[3] = 0;
    v4i_ g3 = {0, 0, 0, 0};
#if __clang_major__ >= 23
    v8i_ g4 = {0, 0, 0, 0, 0, 0, 0, 0};
    __builtin_amdgcn_tensor_load_to_lds(g0, g1, g2, g3, g4, 0);
#else
    __builtin_amdgcn_tensor_load_to_lds(g0, g1, g2, g3, 0);
#endif
}

__device__ __forceinline__ void cp16e(__bf16* __restrict__ d, const __bf16* __restrict__ s) {
    ((uint4*)d)[0] = ((const uint4*)s)[0];
    ((uint4*)d)[1] = ((const uint4*)s)[1];
}

// build v16bf WMMA fragment from two 16B LDS chunks
__device__ __forceinline__ v16bf frag16(const __bf16* lo, const __bf16* hi) {
    v8bf a = *(const v8bf*)lo, b = *(const v8bf*)hi;
    v16bf r;
    #pragma unroll
    for (int i = 0; i < 8; ++i) { r[i] = a[i]; r[8 + i] = b[i]; }
    return r;
}

// ---------------------------------------------------------------------------
// Fused gates GEMM + LSTM cell.
//   WG tile: M=64 (4 m-tiles), Hcol=64 (4 hcol-tiles).
//   8 waves: whalf=wave>>2 -> m-tiles {2*whalf, 2*whalf+1}; wn=wave&3 -> hcol tile.
//   Each wave accumulates acc[2 m-tiles][4 gates]; epilogue does the cell.
// grid: (H/64=8, B/64=4, 2 dirs)
// ---------------------------------------------------------------------------
__global__ __launch_bounds__(256) void k_gatecell(
    __bf16* __restrict__ acat, const __bf16* __restrict__ wcat,
    const float* __restrict__ bsum, float* __restrict__ cst, int depth)
{
    const int dir = blockIdx.z;
    const int dd  = dir * 2 + depth;
    const __bf16* A  = acat + (size_t)dd * Bb * KCAT;
    const __bf16* Wt = wcat + (size_t)dd * Gg * KCAT;

    const int nBase = blockIdx.x * 64;   // hcol base
    const int mBase = blockIdx.y * 64;

    __shared__ __bf16 ldsA[2][64 * LDA];
    __shared__ __bf16 ldsW[2][256 * LDA];   // 4 gates x 64 hcol rows (TDM-padded)

    const int tid   = threadIdx.x;
    const int lane  = tid & 31;
    const int wave  = tid >> 5;
    const int wn    = wave & 3;
    const int whalf = wave >> 2;

    // A staging map: 64 rows x 64 cols, 16 elems per thread
    const int ar = tid >> 2, ac = (tid & 3) * 16;
    const __bf16* arow  = A  + (size_t)(mBase + ar) * KCAT;
    const __bf16* wtile = Wt + (size_t)nBase * KCAT;       // gate-z handled by dim1_stride

    v8f acc[2][4] = {};

    // prologue: stage chunk 0 into buffer 0
    {
        unsigned ao = lds_off(&ldsA[0][ar * LDA + ac]);
        async_b128(ao,      arow + ac);
        async_b128(ao + 16, arow + ac + 8);
        if (wave == 0) tdm_load_w(lds_off(&ldsW[0][0]), wtile);
    }

    int cur = 0;
    for (int kt = 0; kt < KCAT; kt += KC) {
        wait_async0();                                     // own A-tile DMAs done
        if (wave == 0) __builtin_amdgcn_s_wait_tensorcnt(0); // W-tile TDM done
        __syncthreads();
        if (kt + KC < KCAT) {   // overlap: fetch next chunk while computing this one
            const int nx = cur ^ 1, kn = kt + KC;
            unsigned ao = lds_off(&ldsA[nx][ar * LDA + ac]);
            async_b128(ao,      arow + kn + ac);
            async_b128(ao + 16, arow + kn + ac + 8);
            if (wave == 0) tdm_load_w(lds_off(&ldsW[nx][0]), wtile + kn);
        }

        #pragma unroll
        for (int kk = 0; kk < KC; kk += 32) {
            const int kb = kk + ((lane >> 4) << 4);
            const int ka = kk + ((lane >> 4) << 4);
            v16bf af[2];
            #pragma unroll
            for (int mt = 0; mt < 2; ++mt) {
                const int ml = (whalf * 2 + mt) * 16 + (lane & 15);
                af[mt] = frag16(&ldsA[cur][ml * LDA + ka], &ldsA[cur][ml * LDA + ka + 8]);
            }
            #pragma unroll
            for (int g = 0; g < 4; ++g) {
                const int nl = g * 64 + wn * 16 + (lane & 15);
                v16bf bf = frag16(&ldsW[cur][nl * LDA + kb], &ldsW[cur][nl * LDA + kb + 8]);
                #pragma unroll
                for (int mt = 0; mt < 2; ++mt)
                    acc[mt][g] = __builtin_amdgcn_wmma_f32_16x16x32_bf16(
                        false, af[mt], false, bf, (short)0, acc[mt][g], false, false);
            }
        }
        cur ^= 1;
    }

    // ---- fused LSTM cell epilogue (gate order i,f,g,o) ----
    const int hcol = nBase + wn * 16 + (lane & 15);
    const float bi = bsum[(size_t)dd * Gg + 0 * Hh + hcol];
    const float bf_ = bsum[(size_t)dd * Gg + 1 * Hh + hcol];
    const float bg = bsum[(size_t)dd * Gg + 2 * Hh + hcol];
    const float bo = bsum[(size_t)dd * Gg + 3 * Hh + hcol];
    #pragma unroll
    for (int mt = 0; mt < 2; ++mt) {
        #pragma unroll
        for (int e = 0; e < 8; ++e) {
            const int m = mBase + (whalf * 2 + mt) * 16 + ((lane >> 4) << 3) + e;
            const float gi = acc[mt][0][e] + bi;
            const float gf = acc[mt][1][e] + bf_;
            const float gg = acc[mt][2][e] + bg;
            const float go = acc[mt][3][e] + bo;
            const size_t ci = ((size_t)dd * Bb + m) * Hh + hcol;
            const float c = sigf(gf) * cst[ci] + sigf(gi) * tanhf(gg);
            cst[ci] = c;
            acat[((size_t)dd * Bb + m) * KCAT + 256 + hcol] = (__bf16)(sigf(go) * tanhf(c));
        }
    }
}

// ---------------------------------------------------------------------------
// Sync-staged GEMM core (kept for FC): C(64x128), C[m][n]=sum_k A[m][k]*W[n][k]
// ---------------------------------------------------------------------------
__device__ __forceinline__ void gemm_core_64x128(
    const __bf16* __restrict__ A, int lda,
    const __bf16* __restrict__ Wt, int ldw,
    int Ktot, int mBase, int nBase,
    __bf16* ldsA, __bf16* ldsW, v8f acc[4])
{
    const int tid  = threadIdx.x;
    const int lane = tid & 31;
    const int wave = tid >> 5;
    const int ar = tid >> 2, ac = (tid & 3) * 16;
    const int wr = tid >> 1, wc = (tid & 1) * 32;

    for (int kt = 0; kt < Ktot; kt += KC) {
        __syncthreads();
        cp16e(&ldsA[ar * LDA + ac], &A[(size_t)(mBase + ar) * lda + kt + ac]);
        cp16e(&ldsW[wr * LDA + wc],      &Wt[(size_t)(nBase + wr) * ldw + kt + wc]);
        cp16e(&ldsW[wr * LDA + wc + 16], &Wt[(size_t)(nBase + wr) * ldw + kt + wc + 16]);
        if (kt + KC < Ktot) {
            __builtin_prefetch(&A [(size_t)(mBase + ar) * lda + kt + KC + ac], 0, 1);
            __builtin_prefetch(&Wt[(size_t)(nBase + wr) * ldw + kt + KC + wc], 0, 1);
        }
        __syncthreads();

        #pragma unroll
        for (int kk = 0; kk < KC; kk += 32) {
            const int nl = wave * 16 + (lane & 15);
            const int kb = kk + ((lane >> 4) << 4);
            v16bf bf = frag16(&ldsW[nl * LDA + kb], &ldsW[nl * LDA + kb + 8]);
            const int ka = kk + ((lane >> 4) << 4);
            #pragma unroll
            for (int mt = 0; mt < 4; ++mt) {
                const int ml = mt * 16 + (lane & 15);
                v16bf af = frag16(&ldsA[ml * LDA + ka], &ldsA[ml * LDA + ka + 8]);
                acc[mt] = __builtin_amdgcn_wmma_f32_16x16x32_bf16(
                    false, af, false, bf, (short)0, acc[mt], false, false);
            }
        }
    }
}

// ---------------------------------------------------------------------------
// FC: v = h @ wfc^T + bfc.  depth0 -> bf16 into acat[depth1] x-half.
// depth1 -> f32 y into d_out, and pack next x_t bf16 into acat[depth0] x-half.
// grid: (256/128, 256/64, 2)
// ---------------------------------------------------------------------------
__global__ __launch_bounds__(256) void k_fc(
    __bf16* __restrict__ acat, const __bf16* __restrict__ wfcb,
    const float* __restrict__ bfc_ltr, const float* __restrict__ bfc_rtl,
    float* __restrict__ out, const float* __restrict__ x, int t, int depth)
{
    const int dir = blockIdx.z;
    const int dd  = dir * 2 + depth;
    const __bf16* A  = acat + (size_t)dd * Bb * KCAT + 256;   // h-half as A
    const __bf16* Wt = wfcb + (size_t)dd * Vv * Hh;
    const float*  bf = (dir ? bfc_rtl : bfc_ltr) + (size_t)depth * Vv;

    const int mBase = blockIdx.y * 64;
    const int nBase = blockIdx.x * 128;

    __shared__ __bf16 ldsA[64 * LDA];
    __shared__ __bf16 ldsW[128 * LDA];
    v8f acc[4] = {};

    gemm_core_64x128(A, KCAT, Wt, Hh, Hh, mBase, nBase, ldsA, ldsW, acc);

    const int lane = threadIdx.x & 31;
    const int wave = threadIdx.x >> 5;
    const int n    = nBase + wave * 16 + (lane & 15);
    const float bv = bf[n];
    #pragma unroll
    for (int mt = 0; mt < 4; ++mt) {
        #pragma unroll
        for (int e = 0; e < 8; ++e) {
            const int m   = mBase + mt * 16 + ((lane >> 4) << 3) + e;
            const float v = acc[mt][e] + bv;
            if (depth == 0) {
                acat[((size_t)(dir * 2 + 1) * Bb + m) * KCAT + n] = (__bf16)v;
            } else {
                const int tout = dir ? (Tt + t) : t;
                out[((size_t)m * (2 * Tt) + tout) * Vv + n] = v;
                if (t < Tt - 1) {
                    const int tx = dir ? (Tt - 1 - t) : (t + 1);
                    acat[((size_t)(dir * 2 + 0) * Bb + m) * KCAT + n] =
                        (__bf16)x[((size_t)m * Wlen + tx) * Vv + n];
                }
            }
        }
    }
}

// ---------------------------------------------------------------------------
// one-time prep kernels
// ---------------------------------------------------------------------------
__global__ void k_prep_wcat(const float* __restrict__ wih_l, const float* __restrict__ whh_l,
                            const float* __restrict__ wih_r, const float* __restrict__ whh_r,
                            __bf16* __restrict__ wcat)
{
    const size_t N = (size_t)4 * Gg * KCAT;
    for (size_t i = (size_t)blockIdx.x * blockDim.x + threadIdx.x; i < N;
         i += (size_t)gridDim.x * blockDim.x) {
        const int    k = (int)(i % KCAT);
        const size_t r = i / KCAT;
        const int    g = (int)(r % Gg);
        const int    dd = (int)(r / Gg);
        const int depth = dd & 1, dir = dd >> 1;
        const float* wih = dir ? wih_r : wih_l;
        const float* whh = dir ? whh_r : whh_l;
        const float v = (k < Vv)
            ? wih[((size_t)depth * Gg + g) * Vv + k]
            : whh[((size_t)depth * Gg + g) * Hh + (k - Vv)];
        wcat[i] = (__bf16)v;
    }
}

__global__ void k_prep_wfc(const float* __restrict__ wfc_l, const float* __restrict__ wfc_r,
                           __bf16* __restrict__ wfcb)
{
    const size_t N = (size_t)4 * Vv * Hh;
    for (size_t i = (size_t)blockIdx.x * blockDim.x + threadIdx.x; i < N;
         i += (size_t)gridDim.x * blockDim.x) {
        const size_t e  = i % ((size_t)Vv * Hh);
        const int    dd = (int)(i / ((size_t)Vv * Hh));
        const int depth = dd & 1, dir = dd >> 1;
        const float* w = dir ? wfc_r : wfc_l;
        wfcb[i] = (__bf16)w[(size_t)depth * Vv * Hh + e];
    }
}

__global__ void k_prep_bsum(const float* __restrict__ bih_l, const float* __restrict__ bhh_l,
                            const float* __restrict__ bih_r, const float* __restrict__ bhh_r,
                            float* __restrict__ bsum)
{
    const size_t N = (size_t)4 * Gg;
    for (size_t i = (size_t)blockIdx.x * blockDim.x + threadIdx.x; i < N;
         i += (size_t)gridDim.x * blockDim.x) {
        const int g  = (int)(i % Gg);
        const int dd = (int)(i / Gg);
        const int depth = dd & 1, dir = dd >> 1;
        const float* bi = dir ? bih_r : bih_l;
        const float* bh = dir ? bhh_r : bhh_l;
        bsum[i] = bi[(size_t)depth * Gg + g] + bh[(size_t)depth * Gg + g];
    }
}

__global__ void k_prep_state(float* __restrict__ cst, __bf16* __restrict__ acat,
                             const float* __restrict__ x)
{
    const size_t NC = (size_t)4 * Bb * Hh;
    const size_t NA = (size_t)4 * Bb * KCAT;
    for (size_t i = (size_t)blockIdx.x * blockDim.x + threadIdx.x; i < NC + NA;
         i += (size_t)gridDim.x * blockDim.x) {
        if (i < NC) {
            cst[i] = 0.0f;
        } else {
            const size_t j = i - NC;
            const int    k = (int)(j % KCAT);
            const size_t r = j / KCAT;
            const int    b = (int)(r % Bb);
            const int    dd = (int)(r / Bb);
            const int depth = dd & 1, dir = dd >> 1;
            float v = 0.0f;
            if (depth == 0 && k < Vv) {
                const int t0 = dir ? (Wlen - 1) : 0;
                v = x[((size_t)b * Wlen + t0) * Vv + k];
            }
            acat[j] = (__bf16)v;
        }
    }
}

// ---------------------------------------------------------------------------
extern "C" void kernel_launch(void* const* d_in, const int* in_sizes, int n_in,
                              void* d_out, int out_size, void* d_ws, size_t ws_size,
                              hipStream_t stream)
{
    (void)in_sizes; (void)n_in; (void)out_size; (void)ws_size;
    const float* x     = (const float*)d_in[0];
    const float* wih_l = (const float*)d_in[1];
    const float* whh_l = (const float*)d_in[2];
    const float* bih_l = (const float*)d_in[3];
    const float* bhh_l = (const float*)d_in[4];
    const float* wfc_l = (const float*)d_in[5];
    const float* bfc_l = (const float*)d_in[6];
    const float* wih_r = (const float*)d_in[7];
    const float* whh_r = (const float*)d_in[8];
    const float* bih_r = (const float*)d_in[9];
    const float* bhh_r = (const float*)d_in[10];
    const float* wfc_r = (const float*)d_in[11];
    const float* bfc_r = (const float*)d_in[12];
    float* out = (float*)d_out;

    char* ws = (char*)d_ws;
    __bf16* wcat  = (__bf16*)(ws + OFF_WCAT);
    __bf16* wfcb  = (__bf16*)(ws + OFF_WFC);
    float*  bsum  = (float*) (ws + OFF_BSUM);
    float*  cst   = (float*) (ws + OFF_C);
    __bf16* acat  = (__bf16*)(ws + OFF_ACAT);

    k_prep_wcat <<<2048, 256, 0, stream>>>(wih_l, whh_l, wih_r, whh_r, wcat);
    k_prep_wfc  <<<512,  256, 0, stream>>>(wfc_l, wfc_r, wfcb);
    k_prep_bsum <<<32,   256, 0, stream>>>(bih_l, bhh_l, bih_r, bhh_r, bsum);
    k_prep_state<<<1024, 256, 0, stream>>>(cst, acat, x);

    const dim3 gGC(Hh / 64, Bb / 64, 2);   // (8,4,2)
    const dim3 gFc(Vv / 128, Bb / 64, 2);  // (2,4,2)

    for (int t = 0; t < Tt; ++t) {
        for (int depth = 0; depth < 2; ++depth) {
            k_gatecell<<<gGC, 256, 0, stream>>>(acat, wcat, bsum, cst, depth);
            k_fc      <<<gFc, 256, 0, stream>>>(acat, wfcb, bfc_l, bfc_r, out, x, t, depth);
        }
    }
}
